// MultiScaleTemporalAttention_74311524155945
// MI455X (gfx1250) — matrix operand order repaired and checked by reading, not compiled, not run on hardware
//
#include <hip/hip_runtime.h>

typedef __attribute__((ext_vector_type(16))) __bf16 v16bf;
typedef __attribute__((ext_vector_type(8)))  float  v8f;
typedef __attribute__((ext_vector_type(4)))  unsigned int u4;
typedef __attribute__((ext_vector_type(8)))  unsigned int u8v;
typedef __attribute__((ext_vector_type(4)))  float  f4;
typedef __attribute__((ext_vector_type(4)))  __bf16 bf4;

namespace {

constexpr int B_ = 4, L_ = 1024, H_ = 1024, NH_ = 16, HD_ = 64, ML_ = B_ * L_;

union Frag { v16bf v; u4 q[2]; __bf16 e[16]; };

__device__ __forceinline__ v8f wmma_bf16(const Frag& a, const Frag& b, v8f c) {
  // v_wmma_f32_16x16x32_bf16: D = A(16x32) * B(32x16) + C(16x16 f32)
  return __builtin_amdgcn_wmma_f32_16x16x32_bf16(false, a.v, false, b.v,
                                                 (short)0, c, false, false);
}

// CDNA5 async global->LDS copy, tracked with ASYNCcnt (ISA 15.18.3 op 98).
// LDS aperture: flat addr[31:0] == LDS byte offset, so truncation is valid.
__device__ __forceinline__ void async_load_b128(const void* gptr, void* lptr) {
  unsigned lds = (unsigned)(unsigned long long)lptr;
  unsigned long long ga = (unsigned long long)gptr;
  asm volatile("global_load_async_to_lds_b128 %0, %1, off"
               :: "v"(lds), "v"(ga) : "memory");
}
__device__ __forceinline__ void wait_async0() {
  asm volatile("s_wait_asynccnt 0" ::: "memory");
}

// CDNA5 Tensor Data Mover: DMA one 2D tile (16-bit elems) global -> LDS.
// D# built per ISA ch.8: group0 = {count=1, lds_addr, global_addr, type=2},
// group1 = {data_size=2B, tensor_dim0/1, tile_dim0/1, dim0 stride}, groups
// 2/3 zeroed (dims 2..4 unused). Issue once per workgroup (wave 0); tracked
// by TENSORcnt and waited with the probe-verified s_wait_tensorcnt builtin.
__device__ __forceinline__ void tdm_load_2d(const void* gptr, void* lptr,
                                            unsigned tile_d0, unsigned tile_d1,
                                            unsigned tensor_d0, unsigned tensor_d1,
                                            unsigned stride_e) {
  unsigned long long ga = (unsigned long long)gptr;
  unsigned lds = (unsigned)(unsigned long long)lptr;
  u4 g0;
  g0[0] = 1u;                                   // count=1, user descriptor
  g0[1] = lds;                                  // lds_addr (bytes)
  g0[2] = (unsigned)ga;                         // global_addr[31:0]
  g0[3] = (unsigned)(ga >> 32) | (2u << 30);    // global_addr[56:32] | type=2
  u8v g1;
  g1[0] = 1u << 16;                             // data_size=1 (2 bytes), mask=0
  g1[1] = (tensor_d0 & 0xFFFFu) << 16;          // tensor_dim0[15:0]
  g1[2] = (tensor_d0 >> 16) | ((tensor_d1 & 0xFFFFu) << 16);
  g1[3] = (tensor_d1 >> 16) | (tile_d0 << 16);  // tensor_dim1 hi | tile_dim0
  g1[4] = tile_d1;                              // tile_dim1 (tile_dim2 = 0)
  g1[5] = stride_e;                             // tensor_dim0_stride[31:0]
  g1[6] = 0;                                    // stride hi / dim1_stride
  g1[7] = 0;
  u4 gz;
  gz[0] = 0; gz[1] = 0; gz[2] = 0; gz[3] = 0;   // groups 2/3: dims 2..4 unused
  asm volatile("tensor_load_to_lds %0, %1, %2, %3"
               :: "s"(g0), "s"(g1), "s"(gz), "s"(gz) : "memory");
}

// ---------------- fp32 -> bf16 conversion (x4 vectorized) ----------------
__global__ __launch_bounds__(256)
void cvt4_kernel(const float* __restrict__ s, __bf16* __restrict__ d, int n4) {
  int i = blockIdx.x * 256 + threadIdx.x;
  if (i < n4) {
    f4 v = ((const f4*)s)[i];
    bf4 r;
    r[0] = (__bf16)v[0]; r[1] = (__bf16)v[1];
    r[2] = (__bf16)v[2]; r[3] = (__bf16)v[3];
    ((bf4*)d)[i] = r;
  }
}

// ---------------- bf16 GEMM: C[M,N] = A[M,K] x B[K,N] ----------------
// Block tile 128x128, 8 waves, wave -> 32x64 (2x4 WMMA tiles). K step 32.
// Double-buffered LDS; A tiles via one TDM DMA per tile (wave 0), B tiles
// via pipelined global_load + VGPR transpose (B-frag needs K-contiguous).
__global__ __launch_bounds__(256)
void gemm_bf16_kernel(const __bf16* __restrict__ A, const __bf16* __restrict__ Bm,
                      void* __restrict__ C, int M, int N, int K, int out_bf16) {
  __shared__ __bf16 Al[2][128][32];   // row-major A tiles (TDM destination)
  __shared__ __bf16 Btl[2][128][32];  // B tiles transposed [n][k]

  const int tid = threadIdx.x;
  const int w = tid >> 5, lane = tid & 31, lh = lane & 15, hi = lane >> 4;
  const int wm = w & 3, wn = w >> 2;
  const int m0 = blockIdx.y * 128, n0 = blockIdx.x * 128;

  // per-thread B cooperative-load coordinates (2 chunks of 8 bf16 per tile)
  const int c1 = tid + 256;
  const int kb0 = tid >> 4, nb0 = (tid & 15) * 8;
  const int kb1 = c1 >> 4,  nb1 = (c1 & 15) * 8;

  auto issueA = [&](int bufi, int kt) {  // whole 128x32 tile in one TDM op
    tdm_load_2d(&A[(size_t)m0 * K + kt], &Al[bufi][0][0],
                /*tile*/ 32u, 128u, /*tensor*/ (unsigned)K, (unsigned)M,
                /*stride*/ (unsigned)K);
  };
  auto loadB = [&](int kt, u4& r0, u4& r1) {
    r0 = *(const u4*)&Bm[(size_t)(kt + kb0) * N + n0 + nb0];
    r1 = *(const u4*)&Bm[(size_t)(kt + kb1) * N + n0 + nb1];
  };

  v8f acc[2][4];
  for (int a = 0; a < 2; a++)
    for (int b = 0; b < 4; b++)
      for (int j = 0; j < 8; j++) acc[a][b][j] = 0.f;

  u4 br0, br1;
  if (w == 0) issueA(0, 0);
  loadB(0, br0, br1);

  const int nsteps = K / 32;
  for (int i = 0; i < nsteps; i++) {
    const int cur = i & 1;
    __builtin_amdgcn_s_wait_tensorcnt(0);  // wave0: A tile i landed in LDS
    {  // transpose-store B tile i (consumes pipelined global loads)
      const __bf16* te = (const __bf16*)&br0;
      #pragma unroll
      for (int t = 0; t < 8; t++) Btl[cur][nb0 + t][kb0] = te[t];
      te = (const __bf16*)&br1;
      #pragma unroll
      for (int t = 0; t < 8; t++) Btl[cur][nb1 + t][kb1] = te[t];
    }
    __syncthreads();  // tile i fully visible; tile i-1 reads all retired
    if (i + 1 < nsteps) {  // prefetch tile i+1 under this tile's compute
      if (w == 0) issueA(cur ^ 1, (i + 1) * 32);
      loadB((i + 1) * 32, br0, br1);
    }

    Frag af[2];
    #pragma unroll
    for (int mt = 0; mt < 2; mt++) {
      int row = wm * 32 + mt * 16 + lh;
      af[mt].q[0] = *(const u4*)&Al[cur][row][hi * 8];       // K = g*8 + 0..7
      af[mt].q[1] = *(const u4*)&Al[cur][row][16 + hi * 8];  // K = 16 + g*8 + 0..7
    }
    #pragma unroll
    for (int nt = 0; nt < 4; nt++) {
      Frag bf_;
      int n = wn * 64 + nt * 16 + lh;
      bf_.q[0] = *(const u4*)&Btl[cur][n][hi * 8];           // K = g*8 + 0..7
      bf_.q[1] = *(const u4*)&Btl[cur][n][16 + hi * 8];      // K = 16 + g*8 + 0..7
      #pragma unroll
      for (int mt = 0; mt < 2; mt++)
        acc[mt][nt] = wmma_bf16(af[mt], bf_, acc[mt][nt]);
    }
  }

  #pragma unroll
  for (int mt = 0; mt < 2; mt++)
    #pragma unroll
    for (int nt = 0; nt < 4; nt++)
      #pragma unroll
      for (int j = 0; j < 8; j++) {
        int row = m0 + wm * 32 + mt * 16 + j + hi * 8;
        int col = n0 + wn * 64 + nt * 16 + lh;
        float v = acc[mt][nt][j];
        if (out_bf16) ((__bf16*)C)[(size_t)row * N + col] = (__bf16)v;
        else          ((float*)C)[(size_t)row * N + col] = v;
      }
}

// ---------------- fused flash attention (one branch) ----------------
// grid = B*NH*(L/128), block = 256 (8 waves). Wave w owns 16 query rows.
// Double-buffered K/V blocks; K via per-lane async DMA, V via pipelined
// VGPR transpose.
__global__ __launch_bounds__(256)
void attn_kernel(const __bf16* __restrict__ q, const __bf16* __restrict__ kv,
                 float* __restrict__ out, const float* __restrict__ mixw,
                 const float* __restrict__ decf, int use_bias, int accumulate) {
  __shared__ __bf16 Kl[2][32][64];    // K block row-major [key][d]
  __shared__ __bf16 Vl[2][64][32];    // V block transposed [d][key]
  __shared__ __bf16 Pl[8][16][32];    // per-wave P re-striping scratch

  const int tid = threadIdx.x;
  const int w = tid >> 5, lane = tid & 31, lh = lane & 15, hi = lane >> 4;

  const int bh = blockIdx.x >> 3, qc = blockIdx.x & 7;
  const int b = bh >> 4, h = bh & 15;
  const int qbase = qc * 128 + w * 16;
  const int rowQ0 = b * L_ + qbase;

  const float alpha = 1.f / (1.f + __expf(-mixw[0]));
  const float scale = accumulate ? (1.f - alpha) : alpha;
  const float dc = 1.f - decf[0];
  const float rsq = 0.125f;  // 1/sqrt(hd=64)

  // cooperative K/V chunk coordinates: one 16B chunk per thread per tile
  const int kr = tid >> 3, c8 = (tid & 7) * 8;
  auto issueK = [&](int bufi, int kb) {
    const __bf16* src = kv + (size_t)(b * L_ + kb + kr) * (2 * H_) + h * HD_ + c8;
    async_load_b128(src, &Kl[bufi][kr][c8]);
  };
  auto loadV = [&](int kb, u4& r) {
    r = *(const u4*)(kv + (size_t)(b * L_ + kb + kr) * (2 * H_) + H_ + h * HD_ + c8);
  };

  // Resident Q A-fragments (d = 0..31 and 32..63)
  Frag qa[2];
  {
    const __bf16* qr = q + (size_t)(rowQ0 + lh) * H_ + h * HD_;
    qa[0].q[0] = *(const u4*)(qr + hi * 8);
    qa[0].q[1] = *(const u4*)(qr + 16 + hi * 8);
    qa[1].q[0] = *(const u4*)(qr + 32 + hi * 8);
    qa[1].q[1] = *(const u4*)(qr + 48 + hi * 8);
  }

  v8f o[4];
  for (int t = 0; t < 4; t++)
    for (int j = 0; j < 8; j++) o[t][j] = 0.f;
  float mrun[8], lrun[8];
  for (int j = 0; j < 8; j++) { mrun[j] = -1e30f; lrun[j] = 0.f; }

  u4 vr;
  issueK(0, 0);
  loadV(0, vr);

  const int nsteps = L_ / 32;
  for (int it = 0; it < nsteps; it++) {
    const int kb = it * 32;
    const int cur = it & 1;
    wait_async0();
    {  // transpose-store V block it
      const __bf16* te = (const __bf16*)&vr;
      #pragma unroll
      for (int i = 0; i < 8; i++) Vl[cur][c8 + i][kr] = te[i];
    }
    __syncthreads();
    if (it + 1 < nsteps) {  // prefetch block it+1 under compute
      issueK(cur ^ 1, kb + 32);
      loadV(kb + 32, vr);
    }

    // S (16x32) = Q (16x64) * K^T
    v8f s0, s1;
    for (int j = 0; j < 8; j++) { s0[j] = 0.f; s1[j] = 0.f; }
    Frag bk;
    bk.q[0] = *(const u4*)&Kl[cur][lh][hi * 8];
    bk.q[1] = *(const u4*)&Kl[cur][lh][16 + hi * 8];
    s0 = wmma_bf16(qa[0], bk, s0);
    bk.q[0] = *(const u4*)&Kl[cur][lh][32 + hi * 8];
    bk.q[1] = *(const u4*)&Kl[cur][lh][48 + hi * 8];
    s0 = wmma_bf16(qa[1], bk, s0);
    bk.q[0] = *(const u4*)&Kl[cur][16 + lh][hi * 8];
    bk.q[1] = *(const u4*)&Kl[cur][16 + lh][16 + hi * 8];
    s1 = wmma_bf16(qa[0], bk, s1);
    bk.q[0] = *(const u4*)&Kl[cur][16 + lh][32 + hi * 8];
    bk.q[1] = *(const u4*)&Kl[cur][16 + lh][48 + hi * 8];
    s1 = wmma_bf16(qa[1], bk, s1);

    // online softmax; C layout: row = j + 8*hi, col = lh (s0) / 16+lh (s1)
    #pragma unroll
    for (int j = 0; j < 8; j++) {
      int qpos = qbase + j + hi * 8;
      float f0 = s0[j] * rsq, f1 = s1[j] * rsq;
      if (use_bias) {
        f0 -= fabsf((float)(qpos - (kb + lh))) * dc;
        f1 -= fabsf((float)(qpos - (kb + 16 + lh))) * dc;
      }
      float vmax = fmaxf(f0, f1);
      #pragma unroll
      for (int m = 1; m <= 8; m <<= 1) vmax = fmaxf(vmax, __shfl_xor(vmax, m));
      float mnew = fmaxf(mrun[j], vmax);
      float corr = __expf(mrun[j] - mnew);
      float p0 = __expf(f0 - mnew), p1 = __expf(f1 - mnew);
      float rsum = p0 + p1;
      #pragma unroll
      for (int m = 1; m <= 8; m <<= 1) rsum += __shfl_xor(rsum, m);
      lrun[j] = lrun[j] * corr + rsum;
      mrun[j] = mnew;
      #pragma unroll
      for (int t = 0; t < 4; t++) o[t][j] *= corr;
      int prow = j + hi * 8;  // re-stripe P: C-layout -> A-layout via LDS
      Pl[w][prow][lh]      = (__bf16)p0;
      Pl[w][prow][16 + lh] = (__bf16)p1;
    }

    // O (16x64) += P (16x32) * V (32x64)
    Frag pa;
    pa.q[0] = *(const u4*)&Pl[w][lh][hi * 8];
    pa.q[1] = *(const u4*)&Pl[w][lh][16 + hi * 8];
    #pragma unroll
    for (int t = 0; t < 4; t++) {
      Frag bv;
      bv.q[0] = *(const u4*)&Vl[cur][t * 16 + lh][hi * 8];
      bv.q[1] = *(const u4*)&Vl[cur][t * 16 + lh][16 + hi * 8];
      o[t] = wmma_bf16(pa, bv, o[t]);
    }
  }

  // normalize, scale by branch weight, write/accumulate
  #pragma unroll
  for (int j = 0; j < 8; j++) {
    float inv = 1.f / lrun[j];
    int row = rowQ0 + j + hi * 8;
    #pragma unroll
    for (int t = 0; t < 4; t++) {
      int col = h * HD_ + t * 16 + lh;
      float v = o[t][j] * inv * scale;
      float* dst = out + (size_t)row * H_ + col;
      if (accumulate) *dst = *dst + v;
      else            *dst = v;
    }
  }
}

}  // namespace

extern "C" void kernel_launch(void* const* d_in, const int* in_sizes, int n_in,
                              void* d_out, int out_size, void* d_ws, size_t ws_size,
                              hipStream_t stream) {
  (void)in_sizes; (void)n_in; (void)out_size; (void)ws_size;
  const float* x    = (const float*)d_in[0];
  const float* Wq   = (const float*)d_in[1];
  const float* Wkvs = (const float*)d_in[2];
  const float* Wkvl = (const float*)d_in[3];
  const float* Wo   = (const float*)d_in[4];
  const float* mixw = (const float*)d_in[5];
  const float* decf = (const float*)d_in[6];

  // workspace carve-out (~80 MB total)
  char* p = (char*)d_ws;
  auto take = [&](size_t bytes) { char* r = p; p += (bytes + 255) & ~(size_t)255; return r; };
  __bf16* xb   = (__bf16*)take((size_t)ML_ * H_ * 2);  // x bf16, reused for mixed out
  __bf16* wqb  = (__bf16*)take((size_t)H_ * H_ * 2);
  __bf16* wksb = (__bf16*)take((size_t)H_ * 2 * H_ * 2);
  __bf16* wklb = (__bf16*)take((size_t)H_ * 2 * H_ * 2);
  __bf16* wob  = (__bf16*)take((size_t)H_ * H_ * 2);
  __bf16* qb   = (__bf16*)take((size_t)ML_ * H_ * 2);
  __bf16* kvsb = (__bf16*)take((size_t)ML_ * 2 * H_ * 2);
  __bf16* kvlb = (__bf16*)take((size_t)ML_ * 2 * H_ * 2);
  float*  accb = (float*)take((size_t)ML_ * H_ * 4);

  auto cvt = [&](const float* s, __bf16* d, int n) {
    int n4 = n / 4;
    cvt4_kernel<<<(n4 + 255) / 256, 256, 0, stream>>>(s, d, n4);
  };
  cvt(x, xb, ML_ * H_);
  cvt(Wq, wqb, H_ * H_);
  cvt(Wkvs, wksb, H_ * 2 * H_);
  cvt(Wkvl, wklb, H_ * 2 * H_);
  cvt(Wo, wob, H_ * H_);

  // projections
  gemm_bf16_kernel<<<dim3(H_ / 128, ML_ / 128), 256, 0, stream>>>(xb, wqb, qb, ML_, H_, H_, 1);
  gemm_bf16_kernel<<<dim3(2 * H_ / 128, ML_ / 128), 256, 0, stream>>>(xb, wksb, kvsb, ML_, 2 * H_, H_, 1);
  gemm_bf16_kernel<<<dim3(2 * H_ / 128, ML_ / 128), 256, 0, stream>>>(xb, wklb, kvlb, ML_, 2 * H_, H_, 1);

  // short branch (decay bias, alpha, write) then long branch (1-alpha, accumulate)
  attn_kernel<<<B_ * NH_ * (L_ / 128), 256, 0, stream>>>(qb, kvsb, accb, mixw, decf, 1, 0);
  attn_kernel<<<B_ * NH_ * (L_ / 128), 256, 0, stream>>>(qb, kvlb, accb, mixw, decf, 0, 1);

  // mixed output -> bf16 (reuse xb), final projection to f32 d_out
  cvt(accb, xb, ML_ * H_);
  gemm_bf16_kernel<<<dim3(H_ / 128, ML_ / 128), 256, 0, stream>>>(xb, wob, (float*)d_out, ML_, H_, H_, 0);
}
